// WindowTransformerBlock_33646773796921
// MI455X (gfx1250) — compile-verified, matrix operand order checked by reading, not run on hardware
//
#include <hip/hip_runtime.h>
#include <math.h>

typedef _Float16 half_t;
typedef __attribute__((ext_vector_type(16))) _Float16 v16h;
typedef __attribute__((ext_vector_type(8)))  _Float16 v8h;
typedef __attribute__((ext_vector_type(8)))  float    v8f;
typedef __attribute__((ext_vector_type(4)))  int      int4v;

// Static config (matches reference)
#define Bv     32
#define Hv     56
#define Wv     56
#define Cv     384
#define HEADS  12
#define WSZ    7
#define SHIFTv 3
#define Nv     49          // tokens per window
#define NPAD   64          // padded tokens per window
#define HDv    32          // head dim
#define HIDv   1536
#define TOKENS (Bv * Hv * Wv)           // 100352
#define NWIN   (Bv * 64)                // 2048
#define SCALEv 0.17677669529663687f     // 32^-0.5

// ---------------------------------------------------------------------------
// gfx1250 async global<->LDS path (ASYNCcnt) with guarded fallback.
// Builtin signature (from hipcc diagnostic): pointer params are int4* with
// address spaces (1 = global, 3 = LDS), global address first.
// ---------------------------------------------------------------------------
#if __has_builtin(__builtin_amdgcn_global_load_async_to_lds_b128)
#define HAS_ASYNC_LD 1
#else
#define HAS_ASYNC_LD 0
#endif
#if __has_builtin(__builtin_amdgcn_global_store_async_from_lds_b128)
#define HAS_ASYNC_ST 1
#else
#define HAS_ASYNC_ST 0
#endif

typedef __attribute__((address_space(1))) int4v g_int4;
typedef __attribute__((address_space(3))) int4v l_int4;

__device__ __forceinline__ void wait_async0() {
#if __has_builtin(__builtin_amdgcn_s_wait_asynccnt)
    __builtin_amdgcn_s_wait_asynccnt(0);
#else
    asm volatile("s_wait_asynccnt 0x0" ::: "memory");
#endif
}

// Copy n_vec16 16-byte chunks global -> LDS, cooperatively (256 threads).
__device__ __forceinline__ void tile_load_lds(half_t* lds_dst,
                                              const half_t* gsrc, int n_vec16) {
#if HAS_ASYNC_LD
    g_int4* g = (g_int4*)gsrc;
    l_int4* l = (l_int4*)lds_dst;
    for (int i = threadIdx.x; i < n_vec16; i += 256)
        __builtin_amdgcn_global_load_async_to_lds_b128(g + i, l + i, 0, 0);
    wait_async0();
#else
    const uint4* s = (const uint4*)gsrc;
    uint4* d = (uint4*)lds_dst;
    for (int i = threadIdx.x; i < n_vec16; i += 256) d[i] = s[i];
#endif
}

// ---------------------------------------------------------------------------
// WMMA helpers (gfx1250 wave32, v_wmma_f32_16x16x32_f16)
// A operand (16x32, f16): lane m<16 holds row m, K=kb..kb+7 (e0-7) and
// K=kb+16..kb+23 (e8-15), kb=(lane>>4)*8.  B operand (32x16) has the same
// per-lane striping with lane = column; storing B column-major makes the
// identical contiguous row-load work for both operands.
// ---------------------------------------------------------------------------
__device__ __forceinline__ v16h load_frag_row(const half_t* p, int ldm) {
    int lane = threadIdx.x & 31;
    const half_t* q = p + (lane & 15) * ldm + ((lane >> 4) * 8);
    v8h lo = *(const v8h*)(q);
    v8h hi = *(const v8h*)(q + 16);
    v16h out;
#pragma unroll
    for (int e = 0; e < 8; ++e) { out[e] = lo[e]; out[e + 8] = hi[e]; }
    return out;
}

__device__ __forceinline__ v8f wmma16(v16h a, v16h b, v8f c) {
    return __builtin_amdgcn_wmma_f32_16x16x32_f16(false, a, false, b,
                                                  (short)0, c, false, false);
}

// C/D frag layout: lane l, vgpr v -> row = v + (l>>4)*8, col = l&15
__device__ __forceinline__ void store_frag_bias(half_t* p, int ldm, v8f d,
                                                float bias, float scale) {
    int lane = threadIdx.x & 31;
    int col = lane & 15, rb = (lane >> 4) * 8;
#pragma unroll
    for (int v = 0; v < 8; ++v)
        p[(rb + v) * ldm + col] = (half_t)((d[v] + bias) * scale);
}

__device__ __forceinline__ float wave_sum(float v) {
#pragma unroll
    for (int o = 16; o > 0; o >>= 1) v += __shfl_xor(v, o, 32);
    return v;
}
__device__ __forceinline__ float wave_max(float v) {
#pragma unroll
    for (int o = 16; o > 0; o >>= 1) v = fmaxf(v, __shfl_xor(v, o, 32));
    return v;
}

__device__ __forceinline__ int region_of(int p) {
    return p < (Hv - WSZ) ? 0 : (p < (Hv - SHIFTv) ? 1 : 2);
}

// ---------------------------------------------------------------------------
// K0: fp32 -> f16 weight convert with transpose: dst[c*K + k] = src[k*N + c]
// ---------------------------------------------------------------------------
__global__ void cvt_transpose_kernel(const float* __restrict__ s,
                                     half_t* __restrict__ d, int K, int Nn) {
    int i = blockIdx.x * 256 + threadIdx.x;
    if (i < K * Nn) {
        int k = i / Nn, c = i % Nn;
        d[(size_t)c * K + k] = (half_t)s[i];
    }
}

// ---------------------------------------------------------------------------
// K1: LN1 + roll(-3,-3) + window partition -> padded f16 windows
// one window per block; 8 waves x 8 rows; rows 49..63 zero-filled
// ---------------------------------------------------------------------------
__global__ __launch_bounds__(256) void ln_shift_part_kernel(
    const float* __restrict__ x, const float* __restrict__ g,
    const float* __restrict__ b, half_t* __restrict__ xw) {
    int widx = blockIdx.x;
    int bi = widx >> 6, wi = widx & 63, wh = wi >> 3, ww = wi & 7;
    int lane = threadIdx.x & 31, wv = threadIdx.x >> 5;
    half_t* dstw = xw + (size_t)widx * (NPAD * Cv);
    for (int q = 0; q < 8; ++q) {
        int r = wv * 8 + q;
        half_t* dst = dstw + r * Cv;
        if (r >= Nv) {
#pragma unroll
            for (int k = 0; k < 12; ++k) dst[lane + 32 * k] = (half_t)0.f;
            continue;
        }
        int i = r / WSZ, j = r % WSZ;
        int gh = wh * WSZ + i, gw = ww * WSZ + j;
        int sh = (gh + SHIFTv) % Hv, sw = (gw + SHIFTv) % Wv;
        const float* src = x + ((size_t)bi * (Hv * Wv) + sh * Wv + sw) * Cv;
        float vals[12], s = 0.f, ss = 0.f;
#pragma unroll
        for (int k = 0; k < 12; ++k) {
            float v = src[lane + 32 * k];
            vals[k] = v; s += v; ss += v * v;
        }
        s = wave_sum(s); ss = wave_sum(ss);
        float mu = s * (1.f / Cv);
        float rstd = rsqrtf(ss * (1.f / Cv) - mu * mu + 1e-5f);
#pragma unroll
        for (int k = 0; k < 12; ++k) {
            int c = lane + 32 * k;
            dst[c] = (half_t)((vals[k] - mu) * rstd * g[c] + b[c]);
        }
    }
}

// ---------------------------------------------------------------------------
// K2: per-window attention: QKV WMMA -> q k^T WMMA (+bias+mask) -> softmax
//     -> P V WMMA -> staged LDS tile -> contiguous b128 global stores.
// One window per 256-thread block (8 waves). LDS = 88 KB.
// ---------------------------------------------------------------------------
__global__ __launch_bounds__(256) void attn_kernel(
    const half_t* __restrict__ xw, const half_t* __restrict__ wqkvT,
    const float* __restrict__ qkv_b, const float* __restrict__ rel,
    half_t* __restrict__ aout) {
    __shared__ alignas(16) half_t lx[NPAD * Cv];     // 48 KB window activations
    __shared__ alignas(16) half_t qb[NPAD * HDv];    // 4 KB
    __shared__ alignas(16) half_t kb[NPAD * HDv];    // 4 KB
    __shared__ alignas(16) half_t vT[HDv * NPAD];    // 4 KB (transposed V)
    __shared__ alignas(16) half_t ob[NPAD * HDv];    // 4 KB head-output staging
    __shared__ alignas(16) float  sf[NPAD * NPAD];   // 16 KB score scratch
    __shared__ alignas(16) half_t shx[NPAD * NPAD];  // 8 KB softmaxed probs

    int widx = blockIdx.x;
    int wi = widx & 63, wh = wi >> 3, ww = wi & 7;
    int lane = threadIdx.x & 31, wv = threadIdx.x >> 5;

    tile_load_lds(lx, xw + (size_t)widx * (NPAD * Cv), (NPAD * Cv) / 8);
    __syncthreads();

    for (int h = 0; h < HEADS; ++h) {
        // L2->near-cache prefetch of next head's weight columns
        {
            int nh = (h + 1) % HEADS;
            const half_t* wn = wqkvT + (size_t)(nh * HDv) * Cv;
            __builtin_prefetch(wn + (size_t)threadIdx.x * 48, 0, 1);
        }
        // ---- QKV: 24 16x16 tiles (q,k,v x 4x2), K=384 -> 12 WMMA each ----
#pragma unroll
        for (int tt = 0; tt < 3; ++tt) {
            int t = wv * 3 + tt;
            int mat = t >> 3, r8 = t & 7, mt = r8 >> 1, nt = r8 & 1;
            int colbase = mat * Cv + h * HDv + nt * 16;
            v8f acc = {};
#pragma unroll
            for (int kk = 0; kk < 12; ++kk) {
                v16h a = load_frag_row(&lx[(mt * 16) * Cv + kk * 32], Cv);
                v16h bf = load_frag_row(&wqkvT[(size_t)colbase * Cv + kk * 32], Cv);
                acc = wmma16(a, bf, acc);
            }
            float bias = qkv_b[colbase + (lane & 15)];
            if (mat == 0) {
                store_frag_bias(&qb[(mt * 16) * HDv + nt * 16], HDv, acc, bias, SCALEv);
            } else if (mat == 1) {
                store_frag_bias(&kb[(mt * 16) * HDv + nt * 16], HDv, acc, bias, 1.f);
            } else { // V stored transposed: vT[d][m]
                int c = nt * 16 + (lane & 15), rb = (lane >> 4) * 8;
#pragma unroll
                for (int v = 0; v < 8; ++v)
                    vT[c * NPAD + mt * 16 + rb + v] = (half_t)(acc[v] + bias);
            }
        }
        __syncthreads();

        // ---- scores = q k^T : 16 tiles, K=32 (k^T B-frag == row-load of k) ----
#pragma unroll
        for (int tt = 0; tt < 2; ++tt) {
            int t = wv * 2 + tt, mt = t >> 2, nt = t & 3;
            v16h a = load_frag_row(&qb[(mt * 16) * HDv], HDv);
            v16h bf = load_frag_row(&kb[(nt * 16) * HDv], HDv);
            v8f acc = {};
            acc = wmma16(a, bf, acc);
            int m = nt * 16 + (lane & 15);
            int rb = (lane >> 4) * 8;
#pragma unroll
            for (int v = 0; v < 8; ++v) {
                int n = mt * 16 + rb + v;
                float sv;
                if (m >= Nv) {
                    sv = -1e30f; // padded keys never attended
                } else if (n < Nv) {
                    int i1 = n / WSZ, j1 = n % WSZ, i2 = m / WSZ, j2 = m % WSZ;
                    int ridx = (i1 - i2 + WSZ - 1) * (2 * WSZ - 1) + (j1 - j2 + WSZ - 1);
                    float bias = rel[ridx * HEADS + h];
                    int rn = region_of(wh * WSZ + i1) * 3 + region_of(ww * WSZ + j1);
                    int rm = region_of(wh * WSZ + i2) * 3 + region_of(ww * WSZ + j2);
                    sv = acc[v] + bias + ((rn == rm) ? 0.f : -100.f);
                } else {
                    sv = acc[v]; // padded query rows: value irrelevant
                }
                sf[n * NPAD + m] = sv;
            }
        }
        __syncthreads();

        // ---- softmax: 8 rows per wave, 2 cols per lane ----
#pragma unroll
        for (int q = 0; q < 8; ++q) {
            int r = wv * 8 + q;
            float a0 = sf[r * NPAD + lane], a1 = sf[r * NPAD + lane + 32];
            float mx = wave_max(fmaxf(a0, a1));
            float e0 = __expf(a0 - mx), e1 = __expf(a1 - mx);
            float inv = 1.f / wave_sum(e0 + e1);
            shx[r * NPAD + lane]      = (half_t)(e0 * inv);
            shx[r * NPAD + lane + 32] = (half_t)(e1 * inv);
        }
        __syncthreads();

        // ---- out = P V : 8 tiles (one per wave), K=64 -> 2 WMMA; stage in LDS ----
        {
            int mt = wv >> 1, nt = wv & 1;
            v8f acc = {};
#pragma unroll
            for (int kk = 0; kk < 2; ++kk) {
                v16h a = load_frag_row(&shx[(mt * 16) * NPAD + kk * 32], NPAD);
                v16h bf = load_frag_row(&vT[(nt * 16) * NPAD + kk * 32], NPAD);
                acc = wmma16(a, bf, acc);
            }
            store_frag_bias(&ob[(mt * 16) * HDv + nt * 16], HDv, acc, 0.f, 1.f);
        }
        __syncthreads();

        // ---- contiguous 16B copy of the 64x32 head tile into the window row ----
        {
            half_t* gbase = aout + (size_t)widx * (NPAD * Cv) + h * HDv;
            for (int i = threadIdx.x; i < 256; i += 256) { // 64 rows x 4 chunks
                int row = i >> 2, part = i & 3;
#if HAS_ASYNC_ST
                __builtin_amdgcn_global_store_async_from_lds_b128(
                    (g_int4*)(gbase + (size_t)row * Cv + part * 8),
                    (l_int4*)(&ob[row * HDv + part * 8]), 0, 0);
#else
                *(uint4*)(gbase + (size_t)row * Cv + part * 8) =
                    *(const uint4*)(&ob[row * HDv + part * 8]);
#endif
            }
#if HAS_ASYNC_ST
            wait_async0();
#endif
        }
        __syncthreads();
    }
}

// ---------------------------------------------------------------------------
// K3: proj GEMM + window reverse + roll(+3,+3) + residual -> x1 (fp32)
// ---------------------------------------------------------------------------
__global__ __launch_bounds__(256) void proj_kernel(
    const half_t* __restrict__ aout, const half_t* __restrict__ wprojT,
    const float* __restrict__ pb, const float* __restrict__ x,
    float* __restrict__ x1) {
    __shared__ alignas(16) half_t la[NPAD * Cv];
    int widx = blockIdx.x;
    int bi = widx >> 6, wi = widx & 63, wh = wi >> 3, ww = wi & 7;
    int lane = threadIdx.x & 31, wv = threadIdx.x >> 5;
    tile_load_lds(la, aout + (size_t)widx * (NPAD * Cv), (NPAD * Cv) / 8);
    __syncthreads();
    for (int t = wv; t < 96; t += 8) { // 4 mt x 24 nt tiles
        int mt = t / 24, nt = t % 24;
        v8f acc = {};
#pragma unroll
        for (int kk = 0; kk < 12; ++kk) {
            v16h a = load_frag_row(&la[(mt * 16) * Cv + kk * 32], Cv);
            v16h bf = load_frag_row(&wprojT[(size_t)(nt * 16) * Cv + kk * 32], Cv);
            acc = wmma16(a, bf, acc);
        }
        int col = nt * 16 + (lane & 15);
        int rb = (lane >> 4) * 8;
        float bias = pb[col];
#pragma unroll
        for (int v = 0; v < 8; ++v) {
            int r = mt * 16 + rb + v;
            if (r >= Nv) continue;
            int i = r / WSZ, j = r % WSZ;
            int gh = wh * WSZ + i, gw = ww * WSZ + j;
            int dh = (gh + SHIFTv) % Hv, dw = (gw + SHIFTv) % Wv;
            size_t tok = (size_t)bi * (Hv * Wv) + dh * Wv + dw;
            x1[tok * Cv + col] = x[tok * Cv + col] + acc[v] + bias;
        }
    }
}

// ---------------------------------------------------------------------------
// K4: LN2 over tokens -> f16 (one wave per token)
// ---------------------------------------------------------------------------
__global__ __launch_bounds__(256) void ln_tokens_kernel(
    const float* __restrict__ x, const float* __restrict__ g,
    const float* __restrict__ b, half_t* __restrict__ y) {
    int tok = blockIdx.x * 8 + (threadIdx.x >> 5);
    int lane = threadIdx.x & 31;
    const float* src = x + (size_t)tok * Cv;
    half_t* dst = y + (size_t)tok * Cv;
    float vals[12], s = 0.f, ss = 0.f;
#pragma unroll
    for (int k = 0; k < 12; ++k) {
        float v = src[lane + 32 * k];
        vals[k] = v; s += v; ss += v * v;
    }
    s = wave_sum(s); ss = wave_sum(ss);
    float mu = s * (1.f / Cv);
    float rstd = rsqrtf(ss * (1.f / Cv) - mu * mu + 1e-5f);
#pragma unroll
    for (int k = 0; k < 12; ++k) {
        int c = lane + 32 * k;
        dst[c] = (half_t)((vals[k] - mu) * rstd * g[c] + b[c]);
    }
}

// ---------------------------------------------------------------------------
// K5: fused MLP: FC1 + exact GELU -> LDS chunk -> FC2 accumulate; + residual.
// 64 rows per block; hidden (1536) processed in 4 chunks of 384 (no HBM
// round-trip for the 308 MB hidden state).
// ---------------------------------------------------------------------------
__global__ __launch_bounds__(256) void mlp_kernel(
    const half_t* __restrict__ y, const half_t* __restrict__ w1T,
    const float* __restrict__ b1, const half_t* __restrict__ w2T,
    const float* __restrict__ b2, const float* __restrict__ x1,
    float* __restrict__ out) {
    __shared__ alignas(16) half_t ly[64 * Cv];   // 48 KB
    __shared__ alignas(16) half_t lh[64 * Cv];   // 48 KB hidden chunk (gelu'd)
    int rowbase = blockIdx.x * 64;
    int lane = threadIdx.x & 31, wv = threadIdx.x >> 5;
    tile_load_lds(ly, y + (size_t)rowbase * Cv, (64 * Cv) / 8);
    __syncthreads();

    v8f oacc[12];
#pragma unroll
    for (int i = 0; i < 12; ++i) oacc[i] = (v8f){};

    for (int ck = 0; ck < 4; ++ck) {
        // prefetch next chunk's FC1 weight columns out of L2
        {
            int nc = (ck + 1) & 3;
            __builtin_prefetch(w1T + (size_t)(nc * 384) * Cv +
                               (size_t)threadIdx.x * 576, 0, 1);
        }
        // FC1 chunk: 96 tiles / 8 waves, K=384
#pragma unroll
        for (int tt = 0; tt < 12; ++tt) {
            int t = wv * 12 + tt;
            int mt = t / 24, ntc = t % 24;
            v8f acc = {};
#pragma unroll
            for (int kk = 0; kk < 12; ++kk) {
                v16h a = load_frag_row(&ly[(mt * 16) * Cv + kk * 32], Cv);
                v16h bf = load_frag_row(
                    &w1T[(size_t)(ck * 384 + ntc * 16) * Cv + kk * 32], Cv);
                acc = wmma16(a, bf, acc);
            }
            int col = ntc * 16 + (lane & 15);
            float bias = b1[ck * 384 + col];
            int rb = (lane >> 4) * 8;
#pragma unroll
            for (int v = 0; v < 8; ++v) {
                float hv = acc[v] + bias;
                hv = 0.5f * hv * (1.f + erff(hv * 0.70710678118654752f));
                lh[(mt * 16 + rb + v) * Cv + col] = (half_t)hv;
            }
        }
        __syncthreads();
        // FC2 accumulate from this chunk: 96 tiles / 8 waves
#pragma unroll
        for (int tt = 0; tt < 12; ++tt) {
            int t = wv * 12 + tt;
            int mt = t / 24, nt = t % 24;
            v8f acc = oacc[tt];
#pragma unroll
            for (int kk = 0; kk < 12; ++kk) {
                v16h a = load_frag_row(&lh[(mt * 16) * Cv + kk * 32], Cv);
                v16h bf = load_frag_row(
                    &w2T[(size_t)(nt * 16) * HIDv + ck * 384 + kk * 32], HIDv);
                acc = wmma16(a, bf, acc);
            }
            oacc[tt] = acc;
        }
        __syncthreads();
    }

#pragma unroll
    for (int tt = 0; tt < 12; ++tt) {
        int t = wv * 12 + tt;
        int mt = t / 24, nt = t % 24;
        int col = nt * 16 + (lane & 15);
        int rb = (lane >> 4) * 8;
        float bias = b2[col];
#pragma unroll
        for (int v = 0; v < 8; ++v) {
            size_t row = (size_t)rowbase + mt * 16 + rb + v;
            out[row * Cv + col] = x1[row * Cv + col] + bias + oacc[tt][v];
        }
    }
}

// ---------------------------------------------------------------------------
extern "C" void kernel_launch(void* const* d_in, const int* in_sizes, int n_in,
                              void* d_out, int out_size, void* d_ws, size_t ws_size,
                              hipStream_t stream) {
    (void)in_sizes; (void)n_in; (void)out_size; (void)ws_size;
    const float* x     = (const float*)d_in[0];
    const float* n1g   = (const float*)d_in[1];
    const float* n1b   = (const float*)d_in[2];
    const float* qkvw  = (const float*)d_in[3];
    const float* qkvb  = (const float*)d_in[4];
    const float* projw = (const float*)d_in[5];
    const float* projb = (const float*)d_in[6];
    const float* rel   = (const float*)d_in[7];
    const float* n2g   = (const float*)d_in[8];
    const float* n2b   = (const float*)d_in[9];
    const float* fc1w  = (const float*)d_in[10];
    const float* fc1b  = (const float*)d_in[11];
    const float* fc2w  = (const float*)d_in[12];
    const float* fc2b  = (const float*)d_in[13];
    float* out = (float*)d_out;

    // Workspace carve-up (all offsets 16B aligned)
    char* p = (char*)d_ws;
    half_t* wqkvT = (half_t*)p; p += (size_t)Cv * (3 * Cv) * 2;     // 884736 B
    half_t* wprojT = (half_t*)p; p += (size_t)Cv * Cv * 2;          // 294912 B
    half_t* wfc1T = (half_t*)p; p += (size_t)Cv * HIDv * 2;         // 1179648 B
    half_t* wfc2T = (half_t*)p; p += (size_t)HIDv * Cv * 2;         // 1179648 B
    half_t* xw_h  = (half_t*)p; p += (size_t)NWIN * NPAD * Cv * 2;  // ~100 MB
    half_t* attn_h = (half_t*)p; p += (size_t)NWIN * NPAD * Cv * 2; // ~100 MB
    float*  x1    = (float*)p;  p += (size_t)TOKENS * Cv * 4;       // ~154 MB
    half_t* y_h   = xw_h; // xw dead after attention -> reuse for LN2 output

    // K0: weights -> f16, transposed (B-operand becomes contiguous row loads)
    cvt_transpose_kernel<<<(Cv * 3 * Cv + 255) / 256, 256, 0, stream>>>(qkvw, wqkvT, Cv, 3 * Cv);
    cvt_transpose_kernel<<<(Cv * Cv + 255) / 256, 256, 0, stream>>>(projw, wprojT, Cv, Cv);
    cvt_transpose_kernel<<<(Cv * HIDv + 255) / 256, 256, 0, stream>>>(fc1w, wfc1T, Cv, HIDv);
    cvt_transpose_kernel<<<(HIDv * Cv + 255) / 256, 256, 0, stream>>>(fc2w, wfc2T, HIDv, Cv);

    ln_shift_part_kernel<<<NWIN, 256, 0, stream>>>(x, n1g, n1b, xw_h);
    attn_kernel<<<NWIN, 256, 0, stream>>>(xw_h, wqkvT, qkvb, rel, attn_h);
    proj_kernel<<<NWIN, 256, 0, stream>>>(attn_h, wprojT, projb, x, x1);
    ln_tokens_kernel<<<TOKENS / 8, 256, 0, stream>>>(x1, n2g, n2b, y_h);
    mlp_kernel<<<TOKENS / 64, 256, 0, stream>>>(y_h, wfc1T, fc1b, wfc2T, fc2b, x1, out);
}
